// Build_LFRNet_74629351735438
// MI455X (gfx1250) — matrix operand order repaired and checked by reading, not compiled, not run on hardware
//
#include <hip/hip_runtime.h>

#define HH 384
#define WWID 384
#define HWSZ (HH * WWID)
#define NVIEW 49

typedef __attribute__((ext_vector_type(16))) _Float16 v16h;
typedef __attribute__((ext_vector_type(8)))  float    v8f;

// x-scale for grid_sample coordinate mapping: xs = gw * W/(W-1) - 0.5
#define GS_SX (384.0f / 383.0f)

__device__ __forceinline__ float dxf(int v) { return (float)(3 - (v % 7)); }
__device__ __forceinline__ float dyf(int v) { return (float)(3 - (v / 7)); }

// Bilinear sample with zero-padding outside (matches reference tap() validity logic).
__device__ __forceinline__ float bsample(const float* __restrict__ img, float xs, float ys) {
  float x0f = floorf(xs), y0f = floorf(ys);
  float fx = xs - x0f, fy = ys - y0f;
  int x0 = (int)x0f, y0 = (int)y0f;
  float acc = 0.f;
#pragma unroll
  for (int t = 0; t < 4; ++t) {
    int cx = x0 + (t & 1);
    int cy = y0 + (t >> 1);
    float wt = ((t & 1) ? fx : 1.f - fx) * ((t >> 1) ? fy : 1.f - fy);
    if (cx >= 0 && cx < WWID && cy >= 0 && cy < HH)
      acc += img[cy * WWID + cx] * wt;
  }
  return acc;
}

// ---------------------------------------------------------------------------
// 3x3 SAME conv via WMMA f32<-f16.
// Block = 64 threads (2 waves); each wave owns one 16-pixel tile (16
// consecutive x in one row; the 32-pixel block segment never crosses a row
// since W%32==0). GEMM: M=16 pixels, N=16 couts (zero-padded),
// K = CIN*9 padded to a multiple of 32.
//
// Both operands are staged into LDS in the EXACT WMMA per-lane register
// layout, so the hot loop is just ds_load_b128 pairs + v_wmma per K-chunk.
//   wb[ck][half][co][i]   : B fragment, lane n reads 16 contiguous halfs
//   ai[tile][ck][lane][i] : A fragment, lane n reads 16 contiguous halfs
// Bounds checks / masking / f32->f16 conversion happen once per element in
// rolled staging loops (small static footprint).
// ---------------------------------------------------------------------------
template <int CIN>
__global__ __launch_bounds__(64) void conv3x3_wmma(const float* __restrict__ in,
                                                   const float* __restrict__ w,
                                                   const float* __restrict__ bias,
                                                   const float* __restrict__ mask,
                                                   float* __restrict__ out,
                                                   int cout, int relu, int minv) {
  constexpr int KTOT = CIN * 9;
  constexpr int KPAD = ((KTOT + 31) / 32) * 32;
  constexpr int NCH  = KPAD / 32;

  __shared__ alignas(32) _Float16 wb[NCH * 512];      // [ck][half][co][16]
  __shared__ alignas(32) _Float16 ai[2 * NCH * 512];  // [tile][ck][lane][16]
  __shared__ float biaslds[16];

  // --- bias prefetch: gfx1250 async global->LDS path when available -------
#if __has_builtin(__builtin_amdgcn_global_load_async_to_lds_b32) && \
    __has_builtin(__builtin_amdgcn_s_wait_asynccnt)
  if ((int)threadIdx.x < cout) {
    __builtin_amdgcn_global_load_async_to_lds_b32(
        (__attribute__((address_space(1))) int*)(bias + threadIdx.x),
        (__attribute__((address_space(3))) int*)(biaslds + threadIdx.x),
        0, 0);
  }
#else
  if ((int)threadIdx.x < cout) biaslds[threadIdx.x] = bias[threadIdx.x];
#endif

  // --- weight staging into WMMA B layout ----------------------------------
#pragma unroll 1
  for (int idx = threadIdx.x; idx < NCH * 512; idx += 64) {
    int i  = idx & 15;
    int co = (idx >> 4) & 15;
    int K  = ((idx >> 8) << 4) + i;  // = ck*32 + half*16 + i
    float v = (K < KTOT && co < cout) ? w[co * KTOT + K] : 0.f;
    wb[idx] = (_Float16)v;
  }

  // --- block coordinates (uniform / scalar) -------------------------------
  const int pblk   = blockIdx.x * 32;
  const int img    = pblk / HWSZ;
  const int within = pblk % HWSZ;
  const int y      = within / WWID;
  const int x0b    = within % WWID;
  const float* imgbase = in + (size_t)img * CIN * HWSZ;
  __builtin_prefetch(imgbase + y * WWID + x0b, 0, 1);

  // --- im2col staging into WMMA A layout ----------------------------------
#pragma unroll 1
  for (int idx = threadIdx.x; idx < 2 * NCH * 512; idx += 64) {
    int i    = idx & 15;
    int lane = (idx >> 4) & 31;
    int rem  = idx % (NCH * 512);
    int ck   = rem >> 9;
    int t    = idx / (NCH * 512);
    int mrow = lane & 15;
    int K = ck * 32 + ((lane & 16) ? 16 : 0) + i;
    float val = 0.f;
    if (K < KTOT) {
      int c = K / 9, j = K - 9 * c;
      int ky = j / 3, kx = j - 3 * ky;
      int yy = y + ky - 1;
      int xx = x0b + t * 16 + mrow + kx - 1;
      if (yy >= 0 && yy < HH && xx >= 0 && xx < WWID) {
        val = imgbase[c * HWSZ + yy * WWID + xx];
        if (mask) {
          float m = mask[yy * WWID + xx];
          val *= minv ? (1.f - m) : m;
        }
      }
    }
    ai[idx] = (_Float16)val;
  }

#if __has_builtin(__builtin_amdgcn_global_load_async_to_lds_b32) && \
    __has_builtin(__builtin_amdgcn_s_wait_asynccnt)
  __builtin_amdgcn_s_wait_asynccnt(0);
#endif
  __syncthreads();

  // --- hot loop: ds_load_b128 pairs + v_wmma per chunk ---------------------
  const int lane = threadIdx.x & 31;
  const int wave = threadIdx.x >> 5;
  const int nco  = lane & 15;
  const _Float16* ap = ai + (size_t)wave * NCH * 512 + lane * 16;
  const _Float16* bp = wb + (lane >> 4) * 256 + nco * 16;

  v8f acc = {};
#pragma unroll
  for (int ck = 0; ck < NCH; ++ck) {
    v16h a = *(const v16h*)(ap + ck * 512);
    v16h b = *(const v16h*)(bp + ck * 512);
    acc = __builtin_amdgcn_wmma_f32_16x16x32_f16(false, a, false, b,
                                                 (short)0, acc, false, false);
  }

  // --- epilogue: lane owns 8 consecutive pixels of one cout ---------------
  if (nco < cout) {
    float bv = biaslds[nco];
    float r[8];
#pragma unroll
    for (int k = 0; k < 8; ++k) {
      r[k] = acc[k] + bv;
      if (relu) r[k] = fmaxf(r[k], 0.f);
    }
    const int x0 = x0b + wave * 16;
    const int mo = (lane < 16) ? 0 : 8;
    float* obase = out + (size_t)img * cout * HWSZ + (size_t)nco * HWSZ +
                   y * WWID + x0 + mo;
    *(float4*)(obase + 0) = make_float4(r[0], r[1], r[2], r[3]);
    *(float4*)(obase + 4) = make_float4(r[4], r[5], r[6], r[7]);
  }
}

// ---------------------------------------------------------------------------
// Elementwise / warp / splat kernels
// ---------------------------------------------------------------------------
__global__ void minmax_kernel(const float* __restrict__ d, float* __restrict__ mm) {
  __shared__ float smn[256], smx[256];
  int t = threadIdx.x;
  float mn = 3.4e38f, mx = -3.4e38f;
  for (int i = t; i < HWSZ; i += 256) {
    float v = d[i];
    mn = fminf(mn, v);
    mx = fmaxf(mx, v);
  }
  smn[t] = mn; smx[t] = mx;
  __syncthreads();
  for (int s = 128; s > 0; s >>= 1) {
    if (t < s) { smn[t] = fminf(smn[t], smn[t + s]); smx[t] = fmaxf(smx[t], smx[t + s]); }
    __syncthreads();
  }
  if (t == 0) { mm[0] = smn[0]; mm[1] = smx[0]; }
}

__global__ void mask_flowinput_kernel(const float* __restrict__ depthA,
                                      const float* __restrict__ A,
                                      const float* __restrict__ Bimg,
                                      const float* __restrict__ mm,
                                      float* __restrict__ bg,
                                      float* __restrict__ finp) {
  int p = blockIdx.x * 256 + threadIdx.x;
  if (p >= HWSZ) return;
  float mn = mm[0], mx = mm[1];
  float nd = (depthA[p] - mn) / (mx - mn);
  bg[p] = (nd < 0.6f) ? 1.f : 0.f;
  finp[p] = A[24 * HWSZ + p];       // central_saiA (cind = 24)
  finp[HWSZ + p] = Bimg[p];
}

__global__ void warpstack_kernel(const float* __restrict__ A,
                                 const float* __restrict__ depthA,
                                 float* __restrict__ out) {
  int idx = blockIdx.x * 256 + threadIdx.x;
  if (idx >= NVIEW * HWSZ) return;
  int v = idx / HWSZ, p = idx % HWSZ;
  int y = p / WWID, x = p % WWID;
  float d = depthA[p];
  float gw = (float)x + d * dxf(v);
  float gh = (float)y + d * dyf(v);
  out[idx] = bsample(A + (size_t)v * HWSZ, gw * GS_SX - 0.5f, gh * GS_SX - 0.5f);
}

__global__ void flows_kernel(const float* __restrict__ bg,
                             const float* __restrict__ bgf,
                             const float* __restrict__ fgf,
                             const float* __restrict__ A,
                             const float* __restrict__ Bimg,
                             const float* __restrict__ depthA,
                             float* __restrict__ tfA, float* __restrict__ tfB,
                             float* __restrict__ wsaiB, float* __restrict__ wdB,
                             float* __restrict__ wsaiA) {
  int p = blockIdx.x * 256 + threadIdx.x;
  if (p >= HWSZ) return;
  float bgv = bg[p], fgv = 1.f - bgv;
  float a0 = bgf[0 * HWSZ + p] * bgv + fgf[0 * HWSZ + p] * fgv;
  float a1 = bgf[1 * HWSZ + p] * bgv + fgf[1 * HWSZ + p] * fgv;
  float b0 = bgf[2 * HWSZ + p] * bgv + fgf[2 * HWSZ + p] * fgv;
  float b1 = bgf[3 * HWSZ + p] * bgv + fgf[3 * HWSZ + p] * fgv;
  tfA[p] = a0; tfA[HWSZ + p] = a1;
  tfB[p] = b0; tfB[HWSZ + p] = b1;
  int y = p / WWID, x = p % WWID;
  float xsB = ((float)x + b0 * 100.f) * GS_SX - 0.5f;
  float ysB = ((float)y - b1 * 50.f) * GS_SX - 0.5f;
  wsaiB[p] = bsample(A + 24 * HWSZ, xsB, ysB);
  wdB[p]   = bsample(depthA, xsB, ysB);
  float xsA = ((float)x - a0 * 100.f) * GS_SX - 0.5f;
  float ysA = ((float)y + a1 * 50.f) * GS_SX - 0.5f;
  wsaiA[p] = bsample(Bimg, xsA, ysA);
}

__global__ void warpA_depth_kernel(const float* __restrict__ tfA,
                                   const float* __restrict__ depthB,
                                   float* __restrict__ wdA) {
  int p = blockIdx.x * 256 + threadIdx.x;
  if (p >= HWSZ) return;
  int y = p / WWID, x = p % WWID;
  float xs = ((float)x - tfA[p] * 100.f) * GS_SX - 0.5f;
  float ys = ((float)y + tfA[HWSZ + p] * 50.f) * GS_SX - 0.5f;
  wdA[p] = bsample(depthB, xs, ys);
}

__global__ void zero_kernel(float* __restrict__ buf, long n) {
  long i = (long)blockIdx.x * 256 + threadIdx.x;
  if (i < n) buf[i] = 0.f;
}

__global__ void splat_scatter_kernel(const float* __restrict__ src,
                                     const float* __restrict__ depth,
                                     float* __restrict__ accum) {
  int idx = blockIdx.x * 256 + threadIdx.x;
  if (idx >= NVIEW * HWSZ) return;
  int v = idx / HWSZ, p = idx % HWSZ;
  int y = p / WWID, x = p % WWID;
  float d = depth[p];
  float tx = (float)x + d * dxf(v);
  float ty = (float)y + d * dyf(v);
  float x0f = floorf(tx), y0f = floorf(ty);
  float fx = tx - x0f, fy = ty - y0f;
  int x0 = (int)x0f, y0 = (int)y0f;
  float s = src[p];
  float* num = accum + (size_t)v * 2 * HWSZ;
  float* den = num + HWSZ;
#pragma unroll
  for (int t = 0; t < 4; ++t) {
    int cx = x0 + (t & 1);
    int cy = y0 + (t >> 1);
    float wt = ((t & 1) ? fx : 1.f - fx) * ((t >> 1) ? fy : 1.f - fy);
    if (cx >= 0 && cx < WWID && cy >= 0 && cy < HH) {
      int q = cy * WWID + cx;
      atomicAdd(num + q, s * wt);
      atomicAdd(den + q, wt);
    }
  }
}

__global__ void splat_norm_kernel(const float* __restrict__ accum,
                                  float* __restrict__ out) {
  int idx = blockIdx.x * 256 + threadIdx.x;
  if (idx >= NVIEW * HWSZ) return;
  int v = idx / HWSZ, p = idx % HWSZ;
  const float* num = accum + (size_t)v * 2 * HWSZ;
  float de = num[HWSZ + p];
  if (de == 0.f) de = 1.f;
  out[idx] = num[p] / de;
}

// ---------------------------------------------------------------------------
// Host orchestration
// ---------------------------------------------------------------------------
template <int CIN>
static void run_conv(const float* in, const float* w, const float* b, const float* mask,
                     float* out, int cout, int nimg, int relu, int minv, hipStream_t s) {
  int tiles = nimg * (HWSZ / 16);
  int blocks = tiles / 2;  // exact: HWSZ/16 = 9216 is even
  conv3x3_wmma<CIN><<<blocks, 64, 0, s>>>(in, w, b, mask, out, cout, relu, minv);
}

extern "C" void kernel_launch(void* const* d_in, const int* in_sizes, int n_in,
                              void* d_out, int out_size, void* d_ws, size_t ws_size,
                              hipStream_t stream) {
  const float* A    = (const float*)d_in[0];   // (1,49,H,W)
  const float* Bimg = (const float*)d_in[1];   // (1,1,H,W)
  const float* dw1 = (const float*)d_in[2];  const float* db1 = (const float*)d_in[3];
  const float* dw2 = (const float*)d_in[4];  const float* db2 = (const float*)d_in[5];
  const float* bw1 = (const float*)d_in[6];  const float* bb1 = (const float*)d_in[7];
  const float* bw2 = (const float*)d_in[8];  const float* bb2 = (const float*)d_in[9];
  const float* fw1 = (const float*)d_in[10]; const float* fb1 = (const float*)d_in[11];
  const float* fw2 = (const float*)d_in[12]; const float* fb2 = (const float*)d_in[13];
  const float* ow1 = (const float*)d_in[14]; const float* ob1 = (const float*)d_in[15];
  const float* ow2 = (const float*)d_in[16]; const float* ob2 = (const float*)d_in[17];

  float* out = (float*)d_out;
  float* ws  = (float*)d_ws;

  // Output layout (floats, units of HWSZ)
  float* o_warpstack = out + 0L;               // 49
  float* o_wsaiA     = out + 49L * HWSZ;       // 1
  float* o_wsaiB     = out + 50L * HWSZ;       // 1
  float* o_tfA       = out + 51L * HWSZ;       // 2
  float* o_tfB       = out + 53L * HWSZ;       // 2
  float* o_depthA    = out + 55L * HWSZ;       // 1
  float* o_depthB    = out + 56L * HWSZ;       // 1
  float* o_novelB    = out + 57L * HWSZ;       // 49
  float* o_occB      = out + 106L * HWSZ;      // 49
  float* o_recA      = out + 155L * HWSZ;      // 49
  float* o_occA      = out + 204L * HWSZ;      // 49

  // Workspace layout
  float* w_hid   = ws + 0L;                    // 16*HWSZ hidden activations
  float* w_finp  = ws + 16L * HWSZ;            // 2*HWSZ flow_input
  float* w_bg    = ws + 18L * HWSZ;            // 1*HWSZ bg mask
  float* w_bgf   = ws + 19L * HWSZ;            // 4*HWSZ
  float* w_fgf   = ws + 23L * HWSZ;            // 4*HWSZ
  float* w_wdB   = ws + 27L * HWSZ;            // 1*HWSZ warped_depthB
  float* w_wdA   = ws + 28L * HWSZ;            // 1*HWSZ warped_depthA
  float* w_mm    = ws + 29L * HWSZ;            // 2 floats min/max
  float* w_acc   = ws + 30L * HWSZ;            // 98*HWSZ splat accumulator

  const int gpix  = (HWSZ + 255) / 256;            // 576
  const int gview = (NVIEW * HWSZ + 255) / 256;    // 28224
  const long accN = (long)NVIEW * 2 * HWSZ;
  const int gacc  = (int)((accN + 255) / 256);

  // 1) central_depthA = depthnet(A)
  run_conv<49>(A, dw1, db1, nullptr, w_hid, 16, 1, 1, 0, stream);
  run_conv<16>(w_hid, dw2, db2, nullptr, o_depthA, 1, 1, 0, 0, stream);

  // 2) min/max of depthA -> bg mask + flow_input
  minmax_kernel<<<1, 256, 0, stream>>>(o_depthA, w_mm);
  mask_flowinput_kernel<<<gpix, 256, 0, stream>>>(o_depthA, A, Bimg, w_mm, w_bg, w_finp);

  // 3) warped_img_stackA: per-view backward warp of A by depthA disparity
  warpstack_kernel<<<gview, 256, 0, stream>>>(A, o_depthA, o_warpstack);

  // 4) bg / fg flow nets (masked input on conv1 only)
  run_conv<2>(w_finp, bw1, bb1, w_bg, w_hid, 16, 1, 1, 0, stream);
  run_conv<16>(w_hid, bw2, bb2, nullptr, w_bgf, 4, 1, 0, 0, stream);
  run_conv<2>(w_finp, fw1, fb1, w_bg, w_hid, 16, 1, 1, 1, stream);  // 1-bg
  run_conv<16>(w_hid, fw2, fb2, nullptr, w_fgf, 4, 1, 0, 0, stream);

  // 5) target flows + three grid_samples
  flows_kernel<<<gpix, 256, 0, stream>>>(w_bg, w_bgf, w_fgf, A, Bimg, o_depthA,
                                         o_tfA, o_tfB, o_wsaiB, w_wdB, o_wsaiA);

  // 6) novel_lf_imgB = splat_lf(Bimg, warped_depthB)
  zero_kernel<<<gacc, 256, 0, stream>>>(w_acc, accN);
  splat_scatter_kernel<<<gview, 256, 0, stream>>>(Bimg, w_wdB, w_acc);
  splat_norm_kernel<<<gview, 256, 0, stream>>>(w_acc, o_novelB);

  // 7) OccPred_lf_imgB: per-view occ net (1->16->1)
  for (int v = 0; v < NVIEW; ++v) {
    run_conv<1>(o_novelB + (size_t)v * HWSZ, ow1, ob1, nullptr, w_hid, 16, 1, 1, 0, stream);
    run_conv<16>(w_hid, ow2, ob2, nullptr, o_occB + (size_t)v * HWSZ, 1, 1, 0, 0, stream);
  }

  // 8) central_depthB = depthnet(OccPred_lf_imgB)
  run_conv<49>(o_occB, dw1, db1, nullptr, w_hid, 16, 1, 1, 0, stream);
  run_conv<16>(w_hid, dw2, db2, nullptr, o_depthB, 1, 1, 0, 0, stream);

  // 9) warped_depthA = grid_sample(central_depthB, grid_flowA)
  warpA_depth_kernel<<<gpix, 256, 0, stream>>>(o_tfA, o_depthB, w_wdA);

  // 10) rec_lf_imgA = splat_lf(central_saiA, warped_depthA)
  zero_kernel<<<gacc, 256, 0, stream>>>(w_acc, accN);
  splat_scatter_kernel<<<gview, 256, 0, stream>>>(A + 24L * HWSZ, w_wdA, w_acc);
  splat_norm_kernel<<<gview, 256, 0, stream>>>(w_acc, o_recA);

  // 11) OccPred_lf_imgA: per-view occ net
  for (int v = 0; v < NVIEW; ++v) {
    run_conv<1>(o_recA + (size_t)v * HWSZ, ow1, ob1, nullptr, w_hid, 16, 1, 1, 0, stream);
    run_conv<16>(w_hid, ow2, ob2, nullptr, o_occA + (size_t)v * HWSZ, 1, 1, 0, 0, stream);
  }
}
